// Siglip2EncoderLayer_80195629351438
// MI455X (gfx1250) — compile-verified, matrix operand order checked by reading, not run on hardware
//
#include <hip/hip_runtime.h>
#include <hip/hip_bf16.h>

#define S    4096
#define H    1152
#define NH   16
#define HD   72
#define HDP  96     // head dim padded to 3 WMMA k-steps of 32
#define FF   4304
#define EPS  1e-6f

typedef __attribute__((ext_vector_type(16))) __bf16 v16bf;
typedef __attribute__((ext_vector_type(8)))  __bf16 v8bf;
typedef __attribute__((ext_vector_type(8)))  float  v8f;

// ---------------------------------------------------------------------------
// WMMA helper: D = A(16x32 bf16) * B(32x16 bf16) + C(16x16 f32)
// ---------------------------------------------------------------------------
__device__ __forceinline__ v8f wmma_bf16(v16bf a, v16bf b, v8f c) {
    return __builtin_amdgcn_wmma_f32_16x16x32_bf16(
        /*neg_a=*/false, a, /*neg_b=*/false, b,
        /*c_mod=*/(short)0, c, /*reuse_a=*/false, /*reuse_b=*/false);
}

// Load a 16x32 bf16 fragment from a row-major tile.
// CDNA5 16-bit A/B layout: lane L (r = L&15, half = L>>4):
//   elements 0..7  = row r, k =      half*8 .. half*8+7   (contiguous 16B)
//   elements 8..15 = row r, k = 16 + half*8 .. half*8+7   (contiguous 16B)
__device__ __forceinline__ v16bf load_frag(const __bf16* base, int stride) {
    const int lane  = threadIdx.x & 31;
    const int r     = lane & 15;
    const int half8 = (lane >> 4) << 3;
    const __bf16* p = base + (size_t)r * stride + half8;
    v8bf lo = *(const v8bf*)p;
    v8bf hi = *(const v8bf*)(p + 16);
    return __builtin_shufflevector(lo, hi, 0,1,2,3,4,5,6,7,8,9,10,11,12,13,14,15);
}

// CDNA5 async global->LDS copy (32 bytes per lane as two b128 transfers).
// VDST operand carries the LDS byte address = low 32 bits of the shared-
// aperture flat address (flat->LDS mapping truncates to addr[31:0]).
__device__ __forceinline__ void async_copy_b128x2(const __bf16* g, unsigned lds) {
    asm volatile("global_load_async_to_lds_b128 %0, %2, off\n\t"
                 "global_load_async_to_lds_b128 %1, %3, off"
                 :: "v"(lds), "v"(lds + 16u), "v"(g), "v"(g + 8)
                 : "memory");
}
__device__ __forceinline__ void wait_async0() {
    asm volatile("s_wait_asynccnt 0" ::: "memory");
}

// ---------------------------------------------------------------------------
// Elementwise utility kernels
// ---------------------------------------------------------------------------
__global__ void f32_to_bf16_kernel(const float* __restrict__ in,
                                   __bf16* __restrict__ out, size_t n) {
    size_t i  = (size_t)blockIdx.x * blockDim.x + threadIdx.x;
    size_t st = (size_t)gridDim.x * blockDim.x;
    for (; i < n; i += st) out[i] = (__bf16)in[i];
}

__global__ void zero_kernel(unsigned int* __restrict__ p, size_t nwords) {
    size_t i  = (size_t)blockIdx.x * blockDim.x + threadIdx.x;
    size_t st = (size_t)gridDim.x * blockDim.x;
    for (; i < nwords; i += st) p[i] = 0u;
}

// Per-position document span (doc ids are sorted -> contiguous ranges)
__global__ void doc_range_kernel(const int* __restrict__ doc,
                                 int* __restrict__ ds, int* __restrict__ de) {
    int i = blockIdx.x * blockDim.x + threadIdx.x;
    if (i >= S) return;
    int d = doc[i];
    int lo = 0, hi = S;
    while (lo < hi) { int m = (lo + hi) >> 1; if (doc[m] <  d) lo = m + 1; else hi = m; }
    ds[i] = lo;
    lo = 0; hi = S;
    while (lo < hi) { int m = (lo + hi) >> 1; if (doc[m] <= d) lo = m + 1; else hi = m; }
    de[i] = lo;
}

// ---------------------------------------------------------------------------
// LayerNorm: f32 [S,H] -> bf16 [S,H]
// ---------------------------------------------------------------------------
__global__ __launch_bounds__(256) void ln_kernel(const float* __restrict__ x,
                                                 const float* __restrict__ g,
                                                 const float* __restrict__ b,
                                                 __bf16* __restrict__ out) {
    const int row = blockIdx.x;
    const float* xr = x + (size_t)row * H;
    float s = 0.f, s2 = 0.f;
    for (int i = threadIdx.x; i < H; i += 256) { float v = xr[i]; s += v; s2 += v * v; }
    for (int o = 16; o > 0; o >>= 1) {
        s  += __shfl_xor(s,  o, 32);
        s2 += __shfl_xor(s2, o, 32);
    }
    __shared__ float r1[8], r2[8];
    if ((threadIdx.x & 31) == 0) { r1[threadIdx.x >> 5] = s; r2[threadIdx.x >> 5] = s2; }
    __syncthreads();
    float ts = 0.f, ts2 = 0.f;
    for (int i = 0; i < 8; ++i) { ts += r1[i]; ts2 += r2[i]; }
    const float mean = ts * (1.f / H);
    const float var  = ts2 * (1.f / H) - mean * mean;
    const float rstd = rsqrtf(var + EPS);
    for (int i = threadIdx.x; i < H; i += 256)
        out[(size_t)row * H + i] = (__bf16)((xr[i] - mean) * rstd * g[i] + b[i]);
}

// ---------------------------------------------------------------------------
// Tiled WMMA GEMM:  C[M,N] = A[M,K] (row-major bf16) * Bw[N,K]^T (row-major bf16)
// Workgroup tile 128x128, 8 waves (2m x 4n), each wave 64x32 (4x2 WMMA tiles).
// Double-buffered LDS fed by async global->LDS copies (ASYNCcnt path).
// EPI 0: + bias -> bf16 head-padded [NH][S][HDP]            (Q / K projections)
// EPI 1: + bias -> bf16 head-padded transposed [NH][HDP][S] (V projection)
// EPI 2: + bias + residual -> f32 [M,N]                     (O proj / fc2)
// EPI 3: + bias, tanh-GELU -> bf16 [M,N]                    (fc1)
// ---------------------------------------------------------------------------
template <int EPI>
__global__ __launch_bounds__(256) void gemm_kernel(
    const __bf16* __restrict__ A, const __bf16* __restrict__ Bw,
    const float* __restrict__ bias, const float* __restrict__ resid,
    float* __restrict__ outf, __bf16* __restrict__ outb,
    int M, int N, int K) {
    __shared__ __bf16 As[2][128 * 32];
    __shared__ __bf16 Bs[2][128 * 32];

    const int tid  = threadIdx.x;
    const int m0   = blockIdx.y * 128;
    const int n0   = blockIdx.x * 128;
    const int w    = tid >> 5;
    const int lane = tid & 31;
    const int wm   = (w & 1) * 64;    // wave row block (4 m-tiles)
    const int wn   = (w >> 1) * 32;   // wave col block (2 n-tiles)

    v8f acc[4][2] = {};

    const int ksteps = (K + 31) / 32;
    // Staging: both tiles are 128 rows x 32 k; each thread owns one 16-elem
    // (32-byte) chunk of one row in each tile.
    const int row    = tid >> 1;           // 0..127
    const int half16 = (tid & 1) * 16;
    const size_t aRowBase = (size_t)(m0 + row) * K;
    const int    nB       = n0 + row;
    const size_t bRowBase = (size_t)nB * K;
    const bool   nValid   = (nB < N);

    const unsigned chunk = (unsigned)(row * 32 + half16) * 2u;
    const unsigned ldsA[2] = { (unsigned)(uintptr_t)(&As[0][0]) + chunk,
                               (unsigned)(uintptr_t)(&As[1][0]) + chunk };
    const unsigned ldsB[2] = { (unsigned)(uintptr_t)(&Bs[0][0]) + chunk,
                               (unsigned)(uintptr_t)(&Bs[1][0]) + chunk };

    auto stage = [&](int kt, int bi) {
        const int kA = kt * 32 + half16;
        if (kA + 15 < K) {
            async_copy_b128x2(A + aRowBase + kA, ldsA[bi]);
        } else {  // K edge (fc2): synchronous guarded fill
            __bf16* dst = &As[bi][row * 32 + half16];
            for (int j = 0; j < 16; ++j)
                dst[j] = (kA + j < K) ? A[aRowBase + kA + j] : (__bf16)0.f;
        }
        if (nValid && kA + 15 < K) {
            async_copy_b128x2(Bw + bRowBase + kA, ldsB[bi]);
        } else {  // N edge (fc1) or K edge
            __bf16* dst = &Bs[bi][row * 32 + half16];
            for (int j = 0; j < 16; ++j)
                dst[j] = (nValid && kA + j < K) ? Bw[bRowBase + kA + j] : (__bf16)0.f;
        }
    };

    stage(0, 0);
    wait_async0();
    __syncthreads();

    for (int kt = 0; kt < ksteps; ++kt) {
        const int bi = kt & 1;
        if (kt + 1 < ksteps) stage(kt + 1, bi ^ 1);   // prefetch behind compute

        const __bf16* Ab = &As[bi][0];
        const __bf16* Bb = &Bs[bi][0];
        v16bf af[4], bf[2];
        for (int i = 0; i < 4; ++i) af[i] = load_frag(Ab + (size_t)(wm + i * 16) * 32, 32);
        for (int j = 0; j < 2; ++j) bf[j] = load_frag(Bb + (size_t)(wn + j * 16) * 32, 32);
        for (int i = 0; i < 4; ++i)
            for (int j = 0; j < 2; ++j)
                acc[i][j] = wmma_bf16(af[i], bf[j], acc[i][j]);

        if (kt + 1 < ksteps) wait_async0();
        __syncthreads();
    }

    // Epilogue. C/D layout: lane holds col n = lane&15, rows r + 8*(lane>>4).
    const int col = lane & 15, hi8 = (lane >> 4) * 8;
    for (int ti = 0; ti < 4; ++ti)
    for (int tj = 0; tj < 2; ++tj)
    for (int r = 0; r < 8; ++r) {
        const int m = m0 + wm + ti * 16 + r + hi8;
        const int n = n0 + wn + tj * 16 + col;
        if (n >= N) continue;
        float v = acc[ti][tj][r] + bias[n];
        if (EPI == 0) {
            const int h = n / HD, hd = n % HD;
            outb[((size_t)h * S + m) * HDP + hd] = (__bf16)v;
        } else if (EPI == 1) {
            const int h = n / HD, hd = n % HD;
            outb[((size_t)h * HDP + hd) * S + m] = (__bf16)v;
        } else if (EPI == 2) {
            outf[(size_t)m * N + n] = v + resid[(size_t)m * N + n];
        } else {
            const float u = v;
            const float gl = 0.5f * u * (1.f + tanhf(0.7978845608f * (u + 0.044715f * u * u * u)));
            outb[(size_t)m * N + n] = (__bf16)gl;
        }
    }
}

// ---------------------------------------------------------------------------
// Document-masked flash attention.
// 1 wave per (head, 16-query tile); online softmax over 32-key blocks
// restricted to the query tile's document span. Row-sums of P are computed
// with an extra WMMA against an all-ones B matrix (replaces 32 ds_bpermute
// lane reductions per block).
// Q,K: [NH][S][HDP] bf16 (zero-padded), Vt: [NH][HDP][S] bf16.
// ---------------------------------------------------------------------------
__global__ __launch_bounds__(256) void attn_kernel(
    const __bf16* __restrict__ Q, const __bf16* __restrict__ Kb,
    const __bf16* __restrict__ Vt, const int* __restrict__ doc,
    const int* __restrict__ dstart, const int* __restrict__ dend,
    __bf16* __restrict__ out) {
    __shared__ __bf16 Pbuf[8][16 * 32];   // per-wave P-transpose scratch

    const int tid  = threadIdx.x;
    const int w    = tid >> 5;
    const int lane = tid & 31;
    const int wid  = blockIdx.x * 8 + w;     // 0 .. NH*(S/16)-1
    const int h    = wid >> 8;               // S/16 == 256 tiles per head
    const int q0   = (wid & 255) * 16;
    const int col  = lane & 15, hi8 = (lane >> 4) * 8;
    __bf16* Pw = &Pbuf[w][0];

    v16bf ones;
    for (int i = 0; i < 16; ++i) ones[i] = (__bf16)1.0f;

    // Query fragments: 3 K-dim steps over HDP=96
    v16bf qf[3];
    for (int kk = 0; kk < 3; ++kk)
        qf[kk] = load_frag(Q + ((size_t)h * S + q0) * HDP + kk * 32, HDP);

    int dq[8];
    for (int r = 0; r < 8; ++r) dq[r] = doc[q0 + r + hi8];

    float mrun[8], lrun[8];
    v8f o[6] = {};
    for (int r = 0; r < 8; ++r) { mrun[r] = -1e30f; lrun[r] = 0.f; }

    const int kstart = dstart[q0] & ~31;       // doc span of this query tile
    const int kend   = dend[q0 + 15];
    const float scale = 0.1178511302f;         // 72^-0.5

    for (int kb = kstart; kb < kend; kb += 32) {
        // scores: two 16x16 tiles for keys [kb,kb+16) and [kb+16,kb+32)
        v8f s0 = {}, s1 = {};
        for (int kk = 0; kk < 3; ++kk) {
            v16bf b0 = load_frag(Kb + ((size_t)h * S + kb)      * HDP + kk * 32, HDP);
            v16bf b1 = load_frag(Kb + ((size_t)h * S + kb + 16) * HDP + kk * 32, HDP);
            s0 = wmma_bf16(qf[kk], b0, s0);
            s1 = wmma_bf16(qf[kk], b1, s1);
        }
        const int dk0 = doc[kb + col];
        const int dk1 = doc[kb + 16 + col];

        float corr[8];
        for (int r = 0; r < 8; ++r) {
            float a  = (dk0 == dq[r]) ? s0[r] * scale : -1e30f;
            float bb = (dk1 == dq[r]) ? s1[r] * scale : -1e30f;
            float rmax = fmaxf(a, bb);
            for (int off = 1; off < 16; off <<= 1)
                rmax = fmaxf(rmax, __shfl_xor(rmax, off, 16));
            float mnew = fmaxf(mrun[r], rmax);
            float c, p0, p1;
            if (rmax <= -5e29f) {              // no valid key in this block/row
                c = 1.f; p0 = 0.f; p1 = 0.f; mnew = mrun[r];
            } else {
                c  = (mrun[r] <= -5e29f) ? 0.f : __expf(mrun[r] - mnew);
                p0 = (a  <= -5e29f) ? 0.f : __expf(a  - mnew);
                p1 = (bb <= -5e29f) ? 0.f : __expf(bb - mnew);
            }
            mrun[r] = mnew;
            corr[r] = c;
            // write P (16 rows x 32 keys) row-major for A-fragment reload
            Pw[(r + hi8) * 32 + col]      = (__bf16)p0;
            Pw[(r + hi8) * 32 + 16 + col] = (__bf16)p1;
        }
        for (int t = 0; t < 6; ++t)
            for (int r = 0; r < 8; ++r) o[t][r] *= corr[r];

        // same-wave LDS ops are in-order (DScnt); just block compiler reorder
        asm volatile("" ::: "memory");
        v16bf pA = load_frag(Pw, 32);
        v8f zsum = {};
        v8f ssum = wmma_bf16(pA, ones, zsum);       // row-sums of P
        for (int t = 0; t < 6; ++t) {
            v16bf vB = load_frag(Vt + ((size_t)h * HDP + t * 16) * S + kb, S);
            o[t] = wmma_bf16(pA, vB, o[t]);
        }
        for (int r = 0; r < 8; ++r)
            lrun[r] = lrun[r] * corr[r] + ssum[r];
    }

    float inv[8];
    for (int r = 0; r < 8; ++r) inv[r] = (lrun[r] > 0.f) ? 1.f / lrun[r] : 0.f;
    for (int t = 0; t < 6; ++t) {
        const int n = t * 16 + col;
        if (n < HD)
            for (int r = 0; r < 8; ++r)
                out[(size_t)(q0 + r + hi8) * H + h * HD + n] = (__bf16)(o[t][r] * inv[r]);
    }
}

// ---------------------------------------------------------------------------
// Host orchestration
// ---------------------------------------------------------------------------
extern "C" void kernel_launch(void* const* d_in, const int* in_sizes, int n_in,
                              void* d_out, int out_size, void* d_ws, size_t ws_size,
                              hipStream_t stream) {
    (void)in_sizes; (void)n_in; (void)out_size; (void)ws_size;
    const float* x_in  = (const float*)d_in[0];
    const int*   doc   = (const int*)  d_in[1];
    const float* q_w   = (const float*)d_in[2];  const float* q_b   = (const float*)d_in[3];
    const float* k_w   = (const float*)d_in[4];  const float* k_b   = (const float*)d_in[5];
    const float* v_w   = (const float*)d_in[6];  const float* v_b   = (const float*)d_in[7];
    const float* o_w   = (const float*)d_in[8];  const float* o_b   = (const float*)d_in[9];
    const float* ln1_g = (const float*)d_in[10]; const float* ln1_b = (const float*)d_in[11];
    const float* ln2_g = (const float*)d_in[12]; const float* ln2_b = (const float*)d_in[13];
    const float* fc1_w = (const float*)d_in[14]; const float* fc1_b = (const float*)d_in[15];
    const float* fc2_w = (const float*)d_in[16]; const float* fc2_b = (const float*)d_in[17];
    float* out = (float*)d_out;

    // Workspace carve-up (~126 MiB total)
    char* ws = (char*)d_ws;
    size_t off = 0;
    auto carve = [&](size_t bytes) -> void* {
        void* p = ws + off;
        off += (bytes + 255) & ~(size_t)255;
        return p;
    };
    __bf16* WQ  = (__bf16*)carve((size_t)H * H * 2);
    __bf16* WK  = (__bf16*)carve((size_t)H * H * 2);
    __bf16* WV  = (__bf16*)carve((size_t)H * H * 2);
    __bf16* WO  = (__bf16*)carve((size_t)H * H * 2);
    __bf16* WF1 = (__bf16*)carve((size_t)FF * H * 2);
    __bf16* WF2 = (__bf16*)carve((size_t)H * FF * 2);
    __bf16* XLN = (__bf16*)carve((size_t)S * H * 2);          // ln1 out; reused as attn out
    __bf16* QB  = (__bf16*)carve((size_t)NH * S * HDP * 2);   // reused as ln2 out
    __bf16* KB  = (__bf16*)carve((size_t)NH * S * HDP * 2);
    __bf16* VT  = (__bf16*)carve((size_t)NH * HDP * S * 2);
    float*  HID = (float*) carve((size_t)S * H * 4);
    __bf16* ACT = (__bf16*)carve((size_t)S * FF * 2);
    int*    DS  = (int*)   carve((size_t)S * 4);
    int*    DE  = (int*)   carve((size_t)S * 4);

    // 1) weights -> bf16
    f32_to_bf16_kernel<<<1024, 256, 0, stream>>>(q_w,   WQ,  (size_t)H * H);
    f32_to_bf16_kernel<<<1024, 256, 0, stream>>>(k_w,   WK,  (size_t)H * H);
    f32_to_bf16_kernel<<<1024, 256, 0, stream>>>(v_w,   WV,  (size_t)H * H);
    f32_to_bf16_kernel<<<1024, 256, 0, stream>>>(o_w,   WO,  (size_t)H * H);
    f32_to_bf16_kernel<<<1024, 256, 0, stream>>>(fc1_w, WF1, (size_t)FF * H);
    f32_to_bf16_kernel<<<1024, 256, 0, stream>>>(fc2_w, WF2, (size_t)H * FF);

    // 2) document spans + zero head-dim padding of Q/K/Vt (contiguous region)
    doc_range_kernel<<<(S + 255) / 256, 256, 0, stream>>>(doc, DS, DE);
    zero_kernel<<<2048, 256, 0, stream>>>((unsigned int*)QB,
                                          (size_t)3 * NH * S * HDP * 2 / 4);

    // 3) LN1
    ln_kernel<<<S, 256, 0, stream>>>(x_in, ln1_g, ln1_b, XLN);

    // 4) QKV projections (tile 128x128)
    dim3 gH((H + 127) / 128, S / 128);
    gemm_kernel<0><<<gH, 256, 0, stream>>>(XLN, WQ, q_b, nullptr, nullptr, QB, S, H, H);
    gemm_kernel<0><<<gH, 256, 0, stream>>>(XLN, WK, k_b, nullptr, nullptr, KB, S, H, H);
    gemm_kernel<1><<<gH, 256, 0, stream>>>(XLN, WV, v_b, nullptr, nullptr, VT, S, H, H);

    // 5) doc-masked attention (NH * S/16 = 4096 waves, 8 waves/block)
    attn_kernel<<<(NH * (S / 16)) / 8, 256, 0, stream>>>(QB, KB, VT, doc, DS, DE, XLN);

    // 6) O projection + residual -> HID (f32)
    gemm_kernel<2><<<gH, 256, 0, stream>>>(XLN, WO, o_b, x_in, HID, nullptr, S, H, H);

    // 7) LN2 (into QB region, now free)
    ln_kernel<<<S, 256, 0, stream>>>(HID, ln2_g, ln2_b, QB);

    // 8) MLP: fc1 + GELU -> ACT, fc2 + residual -> out
    dim3 gF1((FF + 127) / 128, S / 128);
    gemm_kernel<3><<<gF1, 256, 0, stream>>>(QB, WF1, fc1_b, nullptr, nullptr, ACT, S, FF, H);
    gemm_kernel<2><<<gH, 256, 0, stream>>>(ACT, WF2, fc2_b, HID, out, nullptr, S, H, FF);
}
